// SparseAttention_65953517797445
// MI455X (gfx1250) — matrix operand order repaired and checked by reading, not compiled, not run on hardware
//
#include <hip/hip_runtime.h>

typedef __attribute__((ext_vector_type(16))) __bf16 v16bf;
typedef __attribute__((ext_vector_type(8)))  float  v8f;
typedef unsigned int u32x4 __attribute__((ext_vector_type(4)));
typedef int          i32x8 __attribute__((ext_vector_type(8)));
typedef int          i32x4 __attribute__((ext_vector_type(4)));

#define S_LEN  2048
#define BATCH  2
#define HID    1024
#define NHEADS 16
#define DHEAD  64
#define N3H    3072

#if __has_builtin(__builtin_amdgcn_tensor_load_to_lds)
#define USE_TDM 1
#else
#define USE_TDM 0
#endif

__device__ __forceinline__ unsigned short f2bf(float f) {
    unsigned int u = __float_as_uint(f);
    u = u + 0x7FFFu + ((u >> 16) & 1u);        // round-to-nearest-even
    return (unsigned short)(u >> 16);
}

union FragBF { v16bf v; unsigned int u[8]; };

// A-matrix (16x32 bf16) element-pair k offset for vector pair r (0..7), half = lane/16
__device__ __forceinline__ int a_kpair(int r, int half) {
    return (half << 4) + (r << 1);
}

#if USE_TDM
// Issue one TDM 2D tile load (bf16 elements) global -> LDS, with LDS row padding.
// D# layout per CDNA5 ISA ch.8: group0 = {count/lds_addr/global_addr/type},
// group1 = {flags+pad, tensor dims, tile dims, dim0 stride}. Tracked by TENSORcnt.
__device__ __forceinline__ void tdm_load_2d(unsigned int lds_addr, const void* gptr,
                                            unsigned int tile_d0, unsigned int tile_d1,
                                            unsigned int tensor_d0, unsigned int tensor_d1,
                                            unsigned long long d0_stride,
                                            unsigned int pad_int_code, unsigned int pad_amt_code) {
    unsigned long long ga = (unsigned long long)(size_t)gptr;
    u32x4 g0;
    g0.x = 1u;                                                     // count=1 (valid user D#)
    g0.y = lds_addr;                                               // LDS byte address
    g0.z = (unsigned int)ga;                                       // global_addr[31:0]
    g0.w = (unsigned int)((ga >> 32) & 0x01FFFFFFu) | (2u << 30);  // addr[56:32] | type=2
    i32x8 g1;
    g1[0] = (int)((1u << 16)                    // data_size = 1 -> 2-byte elements
                  | (1u << 20)                  // pad_enable
                  | (pad_int_code << 22)        // DWORDs between pads = 2^(code+1)
                  | (pad_amt_code << 25));      // pad DWORDs = code+1
    g1[1] = (int)((tensor_d0 & 0xFFFFu) << 16);                      // tensor_dim0[15:0]
    g1[2] = (int)((tensor_d0 >> 16) | ((tensor_d1 & 0xFFFFu) << 16));// dim0 hi | dim1 lo
    g1[3] = (int)((tensor_d1 >> 16) | ((tile_d0 & 0xFFFFu) << 16));  // dim1 hi | tile_dim0
    g1[4] = (int)(tile_d1 & 0xFFFFu);                                // tile_dim1 (tile_dim2=0)
    g1[5] = (int)(unsigned int)d0_stride;                            // dim0_stride[31:0]
    g1[6] = (int)(unsigned int)((d0_stride >> 32) & 0xFFFFu);        // dim0_stride[47:32]
    g1[7] = 0;
    i32x4 z4 = {0, 0, 0, 0};
#if __clang_major__ >= 23
    i32x8 z8 = {0, 0, 0, 0, 0, 0, 0, 0};
    __builtin_amdgcn_tensor_load_to_lds(g0, g1, z4, z4, z8, 0);
#else
    __builtin_amdgcn_tensor_load_to_lds(g0, g1, z4, z4, 0);
#endif
}
#endif  // USE_TDM

// ---------------- conversion kernels ----------------
__global__ void cvt_f32_bf16(const float* __restrict__ src, unsigned short* __restrict__ dst, int n) {
    int i = blockIdx.x * blockDim.x + threadIdx.x;
    if (i < n) dst[i] = f2bf(src[i]);
}

// hidden_states [S,B,H] fp32 -> x [B,S,H] bf16
__global__ void cvt_x(const float* __restrict__ hs, unsigned short* __restrict__ x) {
    int i = blockIdx.x * blockDim.x + threadIdx.x;           // over B*S*H
    int b = i / (S_LEN * HID);
    int r = i % (S_LEN * HID);
    int s = r / HID, hh = r % HID;
    x[i] = f2bf(hs[(size_t)(s * BATCH + b) * HID + hh]);
}

// ---------------- tiled bf16 WMMA GEMM ----------------
// C[M=BS x NT] = A[BS x 1024] * W[1024 x NT] + bias
// MODE 0: NT=3072, scatter into q/k/v [B,NH,S,D] bf16
// MODE 1: NT=1024, write fp32 d_out in [S,B,H] order
template <int MODE>
__global__ void __launch_bounds__(128)
gemm_bf16(const unsigned short* __restrict__ Abf, const unsigned short* __restrict__ Wbf,
          const float* __restrict__ bias,
          unsigned short* __restrict__ qb, unsigned short* __restrict__ kb,
          unsigned short* __restrict__ vb, float* __restrict__ outp) {
    constexpr int NT = (MODE == 0) ? N3H : HID;
    __shared__ unsigned short At[64 * 40];   // 64 rows x 32 k (+8 pad)
    __shared__ unsigned short Bt[64 * 40];   // 64 n   x 32 k (+8 pad), transposed W tile

    const int tid  = threadIdx.x;
    const int lane = tid & 31, wave = tid >> 5;
    const int half = lane >> 4, ln = lane & 15;
    const int n0 = blockIdx.x * 64, m0 = blockIdx.y * 64;

    const unsigned int* Ww = (const unsigned int*)Wbf;
#if !USE_TDM
    const unsigned int* Aw = (const unsigned int*)Abf;
#endif

    v8f acc[4] = {};

    for (int k0 = 0; k0 < HID; k0 += 32) {
#if USE_TDM
        // A tile 64x32: one TDM DMA, LDS pad 16B per 64B row -> stride 40 ushorts
        if (wave == 0)
            tdm_load_2d((unsigned int)(size_t)(void*)At,
                        Abf + (size_t)m0 * HID + k0,
                        /*tile*/ 32, 64, /*tensor*/ HID, BATCH * S_LEN,
                        /*stride*/ HID, /*pad 2^(3+1)=16 dw*/ 3, /*4 dw*/ 3);
#else
        #pragma unroll
        for (int idx = tid; idx < 64 * 16; idx += 128) {
            int row = idx >> 4, cp = idx & 15;
            unsigned int val = Aw[(size_t)(m0 + row) * (HID / 2) + (k0 >> 1) + cp];
            *(unsigned int*)&At[row * 40 + cp * 2] = val;
        }
#endif
        // stage W tile 32x64 -> transposed Bt[n][k]  (all waves, overlaps the DMA)
        #pragma unroll
        for (int idx = tid; idx < 32 * 32; idx += 128) {
            int kr = idx >> 5, np = idx & 31;
            unsigned int val = Ww[(size_t)(k0 + kr) * (NT / 2) + (n0 >> 1) + np];
            Bt[(np * 2 + 0) * 40 + kr] = (unsigned short)(val & 0xFFFFu);
            Bt[(np * 2 + 1) * 40 + kr] = (unsigned short)(val >> 16);
        }
#if USE_TDM
        if (wave == 0) __builtin_amdgcn_s_wait_tensorcnt(0);
#endif
        __syncthreads();

        FragBF afr;
        const int mrow = wave * 16 + ln;
        #pragma unroll
        for (int r = 0; r < 8; r++)
            afr.u[r] = *(const unsigned int*)&At[mrow * 40 + a_kpair(r, half)];
        #pragma unroll
        for (int j = 0; j < 4; j++) {
            FragBF bfr;
            #pragma unroll
            for (int r = 0; r < 8; r++)
                bfr.u[r] = *(const unsigned int*)&Bt[(j * 16 + ln) * 40 + (half << 4) + r * 2];
            acc[j] = __builtin_amdgcn_wmma_f32_16x16x32_bf16(false, afr.v, false, bfr.v,
                                                             (short)0, acc[j], false, false);
        }
        __syncthreads();
    }

    // epilogue
    #pragma unroll
    for (int j = 0; j < 4; j++) {
        const int gn = n0 + j * 16 + ln;
        const float bv = bias[gn];
        #pragma unroll
        for (int r = 0; r < 8; r++) {
            const int gm = m0 + wave * 16 + r + half * 8;
            const float val = acc[j][r] + bv;
            const int bb = gm >> 11, ss = gm & (S_LEN - 1);
            if constexpr (MODE == 0) {
                const int head = gn / (3 * DHEAD);
                const int w3 = gn % (3 * DHEAD);
                const int part = w3 / DHEAD, d = w3 % DHEAD;
                const size_t di = ((size_t)(bb * NHEADS + head) * S_LEN + ss) * DHEAD + d;
                const unsigned short bvv = f2bf(val);
                if (part == 0) qb[di] = bvv;
                else if (part == 1) kb[di] = bvv;
                else vb[di] = bvv;
            } else {
                outp[(size_t)(ss * BATCH + bb) * HID + gn] = val;
            }
        }
    }
}

// ---------------- fused flash sparse attention ----------------
// grid = (S/64, B*NH), block = 128 (4 waves, each owning 16 Q rows)
__global__ void __launch_bounds__(128)
attn_kernel(const unsigned short* __restrict__ qb, const unsigned short* __restrict__ kb,
            const unsigned short* __restrict__ vb, const int* __restrict__ mask,
            unsigned short* __restrict__ ctx) {
    __shared__ unsigned short Kt[64 * 72];      // [key][dim]  (+8 pad)
    __shared__ unsigned short Vt[64 * 72];      // [dim][key]  (+8 pad, transposed)
    __shared__ unsigned short Pb[4 * 16 * 72];  // per-wave P staging [m][key]

    const int tid  = threadIdx.x;
    const int lane = tid & 31, wave = tid >> 5;
    const int half = lane >> 4, ln = lane & 15;
    const int qt = blockIdx.x;
    const int bh = blockIdx.y;
    const int b = bh / NHEADS, h = bh % NHEADS;
    const size_t headbase = (size_t)bh * S_LEN * DHEAD;   // elements

    const unsigned int* qw = (const unsigned int*)qb;
    const unsigned int* vw = (const unsigned int*)vb;
#if !USE_TDM
    const unsigned int* kw = (const unsigned int*)kb;
#endif

    const int q0 = qt * 64 + wave * 16;
    const int qrow = q0 + ln;

    // Q fragments for full D=64 (two 16x32 A fragments), resident all iterations
    FragBF qf0, qf1;
    const size_t qbase2 = (headbase + (size_t)qrow * DHEAD) >> 1;
    #pragma unroll
    for (int r = 0; r < 8; r++) {
        const int kp = a_kpair(r, half);
        qf0.u[r] = qw[qbase2 + (kp >> 1)];
        qf1.u[r] = qw[qbase2 + ((kp + 32) >> 1)];
    }

    v8f O[4] = {};
    float mrun[8], lrun[8];
    #pragma unroll
    for (int r = 0; r < 8; r++) { mrun[r] = -1e30f; lrun[r] = 0.0f; }

    unsigned short* Pw = &Pb[wave * 16 * 72];

    for (int kt = 0; kt < S_LEN / 64; kt++) {
        const size_t kbase2 = (headbase + (size_t)kt * 64 * DHEAD) >> 1;
#if USE_TDM
        // K tile 64x64: one TDM DMA, LDS pad 16B per 128B row -> stride 72 ushorts
        if (wave == 0)
            tdm_load_2d((unsigned int)(size_t)(void*)Kt,
                        kb + headbase + (size_t)kt * 64 * DHEAD,
                        /*tile*/ 64, 64, /*tensor*/ DHEAD, S_LEN,
                        /*stride*/ DHEAD, /*pad 2^(4+1)=32 dw*/ 4, /*4 dw*/ 3);
#else
        #pragma unroll
        for (int idx = tid; idx < 64 * 32; idx += 128) {
            int row = idx >> 5, cp = idx & 31;
            unsigned int val = kw[kbase2 + row * 32 + cp];
            *(unsigned int*)&Kt[row * 72 + cp * 2] = val;
        }
#endif
        // V tile transposed (needs VALU transpose -> manual staging, overlaps the DMA)
        #pragma unroll
        for (int idx = tid; idx < 64 * 32; idx += 128) {
            int key = idx >> 5, dp = idx & 31;
            unsigned int val = vw[kbase2 + key * 32 + dp];
            Vt[(dp * 2 + 0) * 72 + key] = (unsigned short)(val & 0xFFFFu);
            Vt[(dp * 2 + 1) * 72 + key] = (unsigned short)(val >> 16);
        }
#if USE_TDM
        if (wave == 0) __builtin_amdgcn_s_wait_tensorcnt(0);
#endif
        __syncthreads();

        // ---- scores: S = (Q K^T) * scale, then mask ----
        float sc[4][8];
        #pragma unroll
        for (int j = 0; j < 4; j++) {
            FragBF kf0, kf1;
            const int key = j * 16 + ln;
            #pragma unroll
            for (int r = 0; r < 8; r++) {
                kf0.u[r] = *(const unsigned int*)&Kt[key * 72 + (half << 4) + r * 2];
                kf1.u[r] = *(const unsigned int*)&Kt[key * 72 + 32 + (half << 4) + r * 2];
            }
            v8f a = {};
            a = __builtin_amdgcn_wmma_f32_16x16x32_bf16(false, qf0.v, false, kf0.v, (short)0, a, false, false);
            a = __builtin_amdgcn_wmma_f32_16x16x32_bf16(false, qf1.v, false, kf1.v, (short)0, a, false, false);
            const int gkey = kt * 64 + j * 16 + ln;
            #pragma unroll
            for (int r = 0; r < 8; r++) {
                const int gq = q0 + r + half * 8;
                float sv = a[r] * 0.125f;                       // 1/sqrt(64)
                if (mask[(size_t)gq * S_LEN + gkey] == 0) sv = -1e30f;
                sc[j][r] = sv;
            }
        }

        // ---- online softmax (rows in VGPR index, cols across 16 lanes) ----
        #pragma unroll
        for (int r = 0; r < 8; r++) {
            float t = fmaxf(fmaxf(sc[0][r], sc[1][r]), fmaxf(sc[2][r], sc[3][r]));
            #pragma unroll
            for (int m = 1; m < 16; m <<= 1) t = fmaxf(t, __shfl_xor(t, m, 32));
            const float nm = fmaxf(mrun[r], t);
            const float alpha = __expf(mrun[r] - nm);
            float ps = 0.0f;
            #pragma unroll
            for (int j = 0; j < 4; j++) {
                float p = (sc[j][r] <= -1e29f) ? 0.0f : __expf(sc[j][r] - nm);
                sc[j][r] = p;
                ps += p;
            }
            #pragma unroll
            for (int m = 1; m < 16; m <<= 1) ps += __shfl_xor(ps, m, 32);
            lrun[r] = lrun[r] * alpha + ps;
            mrun[r] = nm;
            #pragma unroll
            for (int j = 0; j < 4; j++) O[j][r] *= alpha;
        }

        // stage P (C layout -> LDS row-major) for A-layout re-read
        #pragma unroll
        for (int j = 0; j < 4; j++)
            #pragma unroll
            for (int r = 0; r < 8; r++)
                Pw[(r + half * 8) * 72 + j * 16 + ln] = f2bf(sc[j][r]);
        __syncthreads();

        // ---- O += P @ V ----
        FragBF pf0, pf1;
        #pragma unroll
        for (int r = 0; r < 8; r++) {
            const int kp = a_kpair(r, half);
            pf0.u[r] = *(const unsigned int*)&Pw[ln * 72 + kp];
            pf1.u[r] = *(const unsigned int*)&Pw[ln * 72 + 32 + kp];
        }
        #pragma unroll
        for (int jd = 0; jd < 4; jd++) {
            FragBF vf0, vf1;
            const int dim = jd * 16 + ln;
            #pragma unroll
            for (int r = 0; r < 8; r++) {
                vf0.u[r] = *(const unsigned int*)&Vt[dim * 72 + (half << 4) + r * 2];
                vf1.u[r] = *(const unsigned int*)&Vt[dim * 72 + 32 + (half << 4) + r * 2];
            }
            O[jd] = __builtin_amdgcn_wmma_f32_16x16x32_bf16(false, pf0.v, false, vf0.v, (short)0, O[jd], false, false);
            O[jd] = __builtin_amdgcn_wmma_f32_16x16x32_bf16(false, pf1.v, false, vf1.v, (short)0, O[jd], false, false);
        }
        __syncthreads();
    }

    // normalize + store ctx [B,S,H] bf16
    #pragma unroll
    for (int jd = 0; jd < 4; jd++) {
        #pragma unroll
        for (int r = 0; r < 8; r++) {
            const int gq = q0 + r + half * 8;
            const float invl = (lrun[r] > 0.0f) ? (1.0f / lrun[r]) : 0.0f;
            const float val = O[jd][r] * invl;
            const int d = jd * 16 + ln;
            ctx[((size_t)(b * S_LEN + gq)) * HID + h * DHEAD + d] = f2bf(val);
        }
    }
}

// ---------------- host launcher ----------------
extern "C" void kernel_launch(void* const* d_in, const int* in_sizes, int n_in,
                              void* d_out, int out_size, void* d_ws, size_t ws_size,
                              hipStream_t stream) {
    (void)in_sizes; (void)n_in; (void)out_size; (void)ws_size;
    const float* hs   = (const float*)d_in[0];
    const int*   mask = (const int*)d_in[1];
    const float* Wqkv = (const float*)d_in[2];
    const float* bqkv = (const float*)d_in[3];
    const float* Wo   = (const float*)d_in[4];
    const float* bo   = (const float*)d_in[5];
    float* out = (float*)d_out;

    char* ws = (char*)d_ws;
    size_t off = 0;
    auto alloc = [&](size_t bytes) {
        char* p = ws + off;
        off += (bytes + 255) & ~(size_t)255;
        return p;
    };
    unsigned short* xb  = (unsigned short*)alloc((size_t)BATCH * S_LEN * HID * 2);
    unsigned short* wqb = (unsigned short*)alloc((size_t)HID * N3H * 2);
    unsigned short* wob = (unsigned short*)alloc((size_t)HID * HID * 2);
    unsigned short* qb  = (unsigned short*)alloc((size_t)BATCH * NHEADS * S_LEN * DHEAD * 2);
    unsigned short* kb  = (unsigned short*)alloc((size_t)BATCH * NHEADS * S_LEN * DHEAD * 2);
    unsigned short* vb  = (unsigned short*)alloc((size_t)BATCH * NHEADS * S_LEN * DHEAD * 2);
    unsigned short* ctx = (unsigned short*)alloc((size_t)BATCH * S_LEN * HID * 2);

    const int nx = BATCH * S_LEN * HID;
    cvt_x<<<nx / 256, 256, 0, stream>>>(hs, xb);
    cvt_f32_bf16<<<(HID * N3H) / 256, 256, 0, stream>>>(Wqkv, wqb, HID * N3H);
    cvt_f32_bf16<<<(HID * HID) / 256, 256, 0, stream>>>(Wo, wob, HID * HID);

    gemm_bf16<0><<<dim3(N3H / 64, (BATCH * S_LEN) / 64), 128, 0, stream>>>(
        xb, wqb, bqkv, qb, kb, vb, nullptr);

    attn_kernel<<<dim3(S_LEN / 64, BATCH * NHEADS), 128, 0, stream>>>(
        qb, kb, vb, mask, ctx);

    gemm_bf16<1><<<dim3(HID / 64, (BATCH * S_LEN) / 64), 128, 0, stream>>>(
        ctx, wob, bo, nullptr, nullptr, nullptr, out);
}
